// SuperPointNet_28252294873379
// MI455X (gfx1250) — matrix operand order repaired and checked by reading, not compiled, not run on hardware
//
#include <hip/hip_runtime.h>
#include <hip/hip_bf16.h>
#include <math.h>

typedef __attribute__((ext_vector_type(16))) _Float16 v16h;
typedef __attribute__((ext_vector_type(8)))  _Float16 v8h;
typedef __attribute__((ext_vector_type(8)))  float    v8f;

#define HM_H 480
#define HM_W 640
#define NMS_PAD 4
#define TOPK 1024
#define SUPP_W (HM_W + 2*NMS_PAD)
#define SUPP_H (HM_H + 2*NMS_PAD)

// ---------------------------------------------------------------------------
// Weight prepack: W f32 [Cout][Cin][ks][ks] -> f16 A-fragments.
// Fragment f = ((t*(Cin/32) + cc)*(CoutPad/16) + cm), 512 halves per fragment
// laid out [lane][h] so each lane reads its v16h contiguously (2x b128).
// A-matrix 16x32 f16 layout (matches B-fragment order): g=lane>>4, M=lane&15,
//   K(h) = g*16 + h.
// ---------------------------------------------------------------------------
__global__ void prepack_weights(const float* __restrict__ W, _Float16* __restrict__ out,
                                int Cout, int Cin, int ks, int CoutPad) {
  int idx = blockIdx.x * blockDim.x + threadIdx.x;
  int fCm = CoutPad >> 4;
  int nCh = Cin >> 5;
  int total = (ks * ks) * nCh * fCm * 512;
  if (idx >= total) return;
  int h    = idx & 15;
  int lane = (idx >> 4) & 31;
  int f    = idx >> 9;
  int cm = f % fCm;
  int cc = (f / fCm) % nCh;
  int t  = f / (fCm * nCh);
  int g  = lane >> 4;
  int M  = lane & 15;
  int Kl = g * 16 + h;
  int co = cm * 16 + M;
  int ci = cc * 32 + Kl;
  int kh = t / ks, kw = t % ks;
  float v = 0.f;
  if (co < Cout) v = W[((size_t)(co * Cin + ci) * ks + kh) * ks + kw];
  out[idx] = (_Float16)v;
}

// ---------------------------------------------------------------------------
// Implicit-GEMM conv via v_wmma_f32_16x16x32_f16, NHWC activations.
// Block = 256 thr = 8 waves: 4 waves in M (64 Cout) x 2 waves in N, each wave
// holding TWO 16-px accumulators (64 px per block) sharing one A fragment.
// grid = (ceil(W/64), CoutPad/64, H).
// LDS: 2 double-buffered chunks of [KS rows][63+KS cols][32 ch] with the
// per-column stride padded 32->40 halves (20 dwords) so the 16 b128 reads of
// a B-fragment tile all 64 LDS banks conflict-free (20*N mod 64 is a perfect
// 4-dword tiling). Channel-innermost matches the B-fragment order
// (K(h)=(lane>>4)*16+h, N=lane&15) -> 32B-contiguous ds_load_b128 pairs.
// ---------------------------------------------------------------------------
template<int KS>
__global__ __launch_bounds__(256) void conv_wmma(
    const _Float16* __restrict__ in, const _Float16* __restrict__ wfrag,
    const float* __restrict__ bias, _Float16* __restrict__ out,
    int Cin, int Cout, int CoutPad, int H, int W, int relu) {
  constexpr int TW   = 63 + KS;
  constexpr int SEG  = KS * TW;
  constexpr int CSTR = 40;                       // padded channel stride (halves)
  __shared__ __align__(32) _Float16 lds[2][SEG * CSTR];

  const int tid  = threadIdx.x;
  const int lane = tid & 31;
  const int wv   = tid >> 5;
  const int mw   = wv & 3;
  const int nw   = wv >> 2;
  const int xbase  = blockIdx.x * 64;
  const int coBase = blockIdx.y * 64;
  const int y  = blockIdx.z;
  const int y0 = y - KS / 2;
  const int x0 = xbase - KS / 2;
  const int N  = lane & 15;
  const int g  = lane >> 4;
  const int nChunks = Cin >> 5;
  const int cmG = (coBase >> 4) + mw;
  const int fCm = CoutPad >> 4;

  auto fill = [&](int buf, int ci0) {
    for (int e = tid; e < SEG * 4; e += 256) {
      int seg = e >> 2, q = e & 3;
      int r = seg / TW, col = seg - r * TW;
      int yy = y0 + r, xx = x0 + col;
      v8h v = {};
      if (yy >= 0 && yy < H && xx >= 0 && xx < W)
        v = *(const v8h*)&in[((size_t)yy * W + xx) * Cin + ci0 + q * 8];
      *(v8h*)&lds[buf][seg * CSTR + q * 8] = v;
    }
  };

  v8f acc0 = {}, acc1 = {};
  int buf = 0;
  fill(0, 0);
  for (int cc = 0; cc < nChunks; ++cc) {
    __syncthreads();
    if (cc + 1 < nChunks) {
      fill(buf ^ 1, (cc + 1) << 5);
      __builtin_prefetch(wfrag + (size_t)((cc + 1) * fCm + cmG) * 512, 0, 1);
    }
#pragma unroll
    for (int t = 0; t < KS * KS; ++t) {
      const int kh = t / KS, kw = t % KS;
      v16h afrag = *(const v16h*)(wfrag +
          (size_t)((t * nChunks + cc) * fCm + cmG) * 512 + lane * 16);
      const int c0 = nw * 32 + N + kw;
      v16h b0 = *(const v16h*)&lds[buf][(kh * TW + c0) * CSTR + g * 16];
      v16h b1 = *(const v16h*)&lds[buf][(kh * TW + c0 + 16) * CSTR + g * 16];
      acc0 = __builtin_amdgcn_wmma_f32_16x16x32_f16(false, afrag, false, b0,
                                                    (short)0, acc0, false, false);
      acc1 = __builtin_amdgcn_wmma_f32_16x16x32_f16(false, afrag, false, b1,
                                                    (short)0, acc1, false, false);
    }
    buf ^= 1;
  }

  // Epilogue: lane holds 8 consecutive Cout (M = r + 8*(lane>>4)) -> b128 store.
  const int coL = coBase + mw * 16 + 8 * g;
  float bv[8];
#pragma unroll
  for (int r = 0; r < 8; ++r) bv[r] = (coL + r < Cout) ? bias[coL + r] : 0.f;
  const bool vec = ((Cout & 7) == 0) && (coL + 8 <= Cout);
#pragma unroll
  for (int s = 0; s < 2; ++s) {
    int x = xbase + nw * 32 + s * 16 + N;
    if (x >= W) continue;
    v8f a = s ? acc1 : acc0;
    size_t base = ((size_t)y * W + x) * Cout;
    if (vec) {
      v8h hv;
#pragma unroll
      for (int r = 0; r < 8; ++r) {
        float v = a[r] + bv[r];
        if (relu) v = fmaxf(v, 0.f);
        hv[r] = (_Float16)v;
      }
      *(v8h*)&out[base + coL] = hv;
    } else {
#pragma unroll
      for (int r = 0; r < 8; ++r) {
        int co = coL + r;
        if (co < Cout) {
          float v = a[r] + bv[r];
          if (relu) v = fmaxf(v, 0.f);
          out[base + co] = (_Float16)v;
        }
      }
    }
  }
}

// ---------------------------------------------------------------------------
// conv1a: Cin=1 (WMMA would be 1/32 utilized in K) -> direct VALU kernel.
// Output NHWC [H][W][64]; channel-innermost thread mapping for coalescing.
// ---------------------------------------------------------------------------
__global__ void conv1a_kernel(const float* __restrict__ img, const float* __restrict__ w,
                              const float* __restrict__ b, _Float16* __restrict__ out,
                              int H, int W) {
  int idx = blockIdx.x * blockDim.x + threadIdx.x;
  if (idx >= 64 * H * W) return;
  int c = idx & 63;
  int p = idx >> 6;
  int y = p / W, x = p - y * W;
  float s = b[c];
#pragma unroll
  for (int kh = 0; kh < 3; ++kh)
#pragma unroll
    for (int kw = 0; kw < 3; ++kw) {
      int yy = y + kh - 1, xx = x + kw - 1;
      if (yy >= 0 && yy < H && xx >= 0 && xx < W)
        s += w[c * 9 + kh * 3 + kw] * img[yy * W + xx];
    }
  out[(size_t)p * 64 + c] = (_Float16)fmaxf(s, 0.f);
}

// 2x2 maxpool, NHWC.
__global__ void pool_kernel(const _Float16* __restrict__ in, _Float16* __restrict__ out,
                            int C, int H, int W) {
  int Ho = H >> 1, Wo = W >> 1;
  int idx = blockIdx.x * blockDim.x + threadIdx.x;
  if (idx >= C * Ho * Wo) return;
  int c = idx % C;
  int p = idx / C;
  int x = p % Wo, y = p / Wo;
  const _Float16* ip = in + ((size_t)(2 * y) * W + 2 * x) * C + c;
  size_t rs = (size_t)W * C;
  float a = (float)ip[0], bb = (float)ip[C], cc = (float)ip[rs], d = (float)ip[rs + C];
  out[((size_t)y * Wo + x) * C + c] = (_Float16)fmaxf(fmaxf(a, bb), fmaxf(cc, d));
}

// semi f16 NHWC [4800][65] -> heatmap f32 [480][640] (exp/(sum+1e-5), pixel-shuffle)
__global__ void softmax_heat_kernel(const _Float16* __restrict__ semi, float* __restrict__ hm) {
  int idx = blockIdx.x * blockDim.x + threadIdx.x;
  if (idx >= 60 * 80) return;
  const _Float16* sp = semi + (size_t)idx * 65;
  float e[65];
  float s = 0.f;
#pragma unroll
  for (int c = 0; c < 65; ++c) {
    e[c] = expf((float)sp[c]);
    s += e[c];
  }
  float inv = 1.f / (s + 1e-5f);
  int hc = idx / 80, wc = idx - hc * 80;
#pragma unroll
  for (int c = 0; c < 64; ++c) {
    int r = c >> 3, cl = c & 7;
    hm[(size_t)(hc * 8 + r) * HM_W + wc * 8 + cl] = e[c] * inv;
  }
}

// desc16 NHWC [4800][256] -> cd f32 [4800][256], L2-normalized over channels.
__global__ void norm_desc_kernel(const _Float16* __restrict__ d16, float* __restrict__ cd) {
  __shared__ float red[256];
  int p = blockIdx.x;
  int c = threadIdx.x;
  float v = (float)d16[(size_t)p * 256 + c];
  red[c] = v * v;
  __syncthreads();
  for (int s = 128; s > 0; s >>= 1) {
    if (c < s) red[c] += red[c + s];
    __syncthreads();
  }
  float nrm = sqrtf(red[0]);
  cd[(size_t)p * 256 + c] = v / nrm;
}

// Iterative argmax top-K (matches top_k sorted order; ties -> lowest index).
__global__ __launch_bounds__(1024) void topk_kernel(float* __restrict__ hm,
                                                    float* __restrict__ scores,
                                                    int* __restrict__ idxout) {
  __shared__ float sv[1024];
  __shared__ int si[1024];
  const int tid = threadIdx.x;
  for (int k = 0; k < TOPK; ++k) {
    float best = -INFINITY;
    int bi = 0x7fffffff;
    for (int i = tid; i < HM_H * HM_W; i += 1024) {
      float v = hm[i];
      if (v > best || (v == best && i < bi)) { best = v; bi = i; }
    }
    sv[tid] = best; si[tid] = bi;
    __syncthreads();
    for (int s = 512; s > 0; s >>= 1) {
      if (tid < s) {
        float ov = sv[tid + s]; int oi = si[tid + s];
        if (ov > sv[tid] || (ov == sv[tid] && oi < si[tid])) { sv[tid] = ov; si[tid] = oi; }
      }
      __syncthreads();
    }
    if (tid == 0) {
      scores[k] = sv[0];
      idxout[k] = si[0];
      hm[si[0]] = -INFINITY;
    }
    __syncthreads();
  }
}

__global__ void zero_supp_kernel(unsigned char* supp) {
  int i = blockIdx.x * blockDim.x + threadIdx.x;
  if (i < SUPP_H * SUPP_W) supp[i] = 0;
}

// Sequential NMS scan (exactly the reference's lax.scan semantics).
__global__ void nms_kernel(const float* __restrict__ scores, const int* __restrict__ idx,
                           unsigned char* __restrict__ supp, int* __restrict__ keep,
                           float* __restrict__ outKp, float* __restrict__ outKeep) {
  __shared__ int sy, sx, sk;
  for (int k = 0; k < TOPK; ++k) {
    if (threadIdx.x == 0) {
      int id = idx[k];
      int y = id / HM_W;
      int x = id - y * HM_W;
      int valid = scores[k] >= 0.015f;
      int occ = supp[(y + NMS_PAD) * SUPP_W + (x + NMS_PAD)];
      int kp = valid && !occ;
      sy = y; sx = x; sk = kp;
      keep[k] = kp;
      outKp[2 * k]     = (float)x;
      outKp[2 * k + 1] = (float)y;
      outKeep[k] = (float)kp;
    }
    __syncthreads();
    if (sk && threadIdx.x < 81) {
      int r = threadIdx.x / 9, c = threadIdx.x % 9;
      supp[(sy + r) * SUPP_W + (sx + c)] = 1;
    }
    __syncthreads();
  }
}

// Bilinear sample of cd [4800][256] at each keypoint, renormalize, scale by keep.
__global__ void sample_desc_kernel(const float* __restrict__ cd, const int* __restrict__ idx,
                                   const float* __restrict__ keepf, float* __restrict__ outDesc) {
  __shared__ float red[256];
  __shared__ float w4[4];
  __shared__ int i4[4];
  int k = blockIdx.x;
  int c = threadIdx.x;
  if (c == 0) {
    int id = idx[k];
    int ys = id / HM_W;
    int xs = id - ys * HM_W;
    float px = (float)xs * 79.0f / 640.0f;   // xs*(w-1)/W
    float py = (float)ys * 59.0f / 480.0f;   // ys*(h-1)/H
    float x0f = floorf(px), y0f = floorf(py);
    float wx = px - x0f, wy = py - y0f;
    int x0i = min(max((int)x0f, 0), 79);
    int x1i = min(x0i + 1, 79);
    int y0i = min(max((int)y0f, 0), 59);
    int y1i = min(y0i + 1, 59);
    w4[0] = (1.f - wy) * (1.f - wx);
    w4[1] = (1.f - wy) * wx;
    w4[2] = wy * (1.f - wx);
    w4[3] = wy * wx;
    i4[0] = y0i * 80 + x0i;
    i4[1] = y0i * 80 + x1i;
    i4[2] = y1i * 80 + x0i;
    i4[3] = y1i * 80 + x1i;
  }
  __syncthreads();
  float d = cd[(size_t)i4[0] * 256 + c] * w4[0] +
            cd[(size_t)i4[1] * 256 + c] * w4[1] +
            cd[(size_t)i4[2] * 256 + c] * w4[2] +
            cd[(size_t)i4[3] * 256 + c] * w4[3];
  red[c] = d * d;
  __syncthreads();
  for (int s = 128; s > 0; s >>= 1) {
    if (c < s) red[c] += red[c + s];
    __syncthreads();
  }
  float nrm = fmaxf(sqrtf(red[0]), 1e-12f);
  outDesc[(size_t)k * 256 + c] = (d / nrm) * keepf[k];
}

// ---------------------------------------------------------------------------
extern "C" void kernel_launch(void* const* d_in, const int* in_sizes, int n_in,
                              void* d_out, int out_size, void* d_ws, size_t ws_size,
                              hipStream_t stream) {
  (void)in_sizes; (void)n_in; (void)out_size; (void)ws_size;

  char* ws = (char*)d_ws;
  size_t o = 0;
  auto take = [&](size_t bytes) -> char* {
    char* p = ws + o;
    o += (bytes + 255) & ~(size_t)255;
    return p;
  };
  _Float16* actA  = (_Float16*)take((size_t)64 * 480 * 640 * 2);
  _Float16* actB  = (_Float16*)take((size_t)64 * 480 * 640 * 2);
  _Float16* wbuf  = (_Float16*)take((size_t)768 * 1024);      // largest prepack = 576 KiB
  _Float16* semi16 = (_Float16*)take((size_t)4800 * 65 * 2);
  _Float16* desc16 = (_Float16*)take((size_t)4800 * 256 * 2);
  float* hm      = (float*)take((size_t)HM_H * HM_W * 4);
  float* cd      = (float*)take((size_t)4800 * 256 * 4);
  float* scores  = (float*)take(TOPK * 4);
  int*   topidx  = (int*)take(TOPK * 4);
  int*   keepArr = (int*)take(TOPK * 4);
  unsigned char* supp = (unsigned char*)take((size_t)SUPP_H * SUPP_W);

  auto F = [&](int i) { return (const float*)d_in[i]; };

  auto runconv = [&](int wi, int bi, const _Float16* src, _Float16* dst,
                     int Cin, int Cout, int H, int W, int ks, int relu) {
    int CoutPad = ((Cout + 63) / 64) * 64;
    int total = (ks * ks) * (Cin / 32) * (CoutPad / 16) * 512;
    prepack_weights<<<(total + 255) / 256, 256, 0, stream>>>(F(wi), wbuf, Cout, Cin, ks, CoutPad);
    dim3 grid((W + 63) / 64, CoutPad / 64, H);
    if (ks == 3)
      conv_wmma<3><<<grid, 256, 0, stream>>>(src, wbuf, F(bi), dst, Cin, Cout, CoutPad, H, W, relu);
    else
      conv_wmma<1><<<grid, 256, 0, stream>>>(src, wbuf, F(bi), dst, Cin, Cout, CoutPad, H, W, relu);
  };

  // conv1a: image -> actA NHWC [480*640][64]
  {
    int total = 64 * 480 * 640;
    conv1a_kernel<<<(total + 255) / 256, 256, 0, stream>>>(F(0), F(1), F(2), actA, 480, 640);
  }
  runconv(3, 4, actA, actB, 64, 64, 480, 640, 3, 1);             // conv1b
  pool_kernel<<<((64 * 240 * 320) + 255) / 256, 256, 0, stream>>>(actB, actA, 64, 480, 640);
  runconv(5, 6, actA, actB, 64, 64, 240, 320, 3, 1);             // conv2a
  runconv(7, 8, actB, actA, 64, 64, 240, 320, 3, 1);             // conv2b
  pool_kernel<<<((64 * 120 * 160) + 255) / 256, 256, 0, stream>>>(actA, actB, 64, 240, 320);
  runconv(9, 10, actB, actA, 64, 128, 120, 160, 3, 1);           // conv3a
  runconv(11, 12, actA, actB, 128, 128, 120, 160, 3, 1);         // conv3b
  pool_kernel<<<((128 * 60 * 80) + 255) / 256, 256, 0, stream>>>(actB, actA, 128, 120, 160);
  runconv(13, 14, actA, actB, 128, 128, 60, 80, 3, 1);           // conv4a
  runconv(15, 16, actB, actA, 128, 128, 60, 80, 3, 1);           // conv4b -> x in actA
  runconv(17, 18, actA, actB, 128, 256, 60, 80, 3, 1);           // convPa
  runconv(19, 20, actB, semi16, 256, 65, 60, 80, 1, 0);          // convPb
  runconv(21, 22, actA, actB, 128, 256, 60, 80, 3, 1);           // convDa
  runconv(23, 24, actB, desc16, 256, 256, 60, 80, 1, 0);         // convDb

  softmax_heat_kernel<<<(4800 + 127) / 128, 128, 0, stream>>>(semi16, hm);
  norm_desc_kernel<<<4800, 256, 0, stream>>>(desc16, cd);
  topk_kernel<<<1, 1024, 0, stream>>>(hm, scores, topidx);
  zero_supp_kernel<<<((SUPP_H * SUPP_W) + 255) / 256, 256, 0, stream>>>(supp);

  float* outKp   = (float*)d_out;            // [1024][2]
  float* outDesc = outKp + 2 * TOPK;         // [1024][256]
  float* outKeep = outDesc + 256 * TOPK;     // [1024]
  nms_kernel<<<1, 128, 0, stream>>>(scores, topidx, supp, keepArr, outKp, outKeep);
  sample_desc_kernel<<<TOPK, 256, 0, stream>>>(cd, topidx, outKeep, outDesc);
}
